// Attention_13718125543518
// MI455X (gfx1250) — compile-verified, matrix operand order checked by reading, not run on hardware
//
#include <hip/hip_runtime.h>
#include <hip/hip_bf16.h>
#include <stdint.h>

// Problem constants (match reference)
#define B_   16
#define N_   1024
#define DIM_ 512
#define H_   8
#define DH_  64
#define SCALE_ 0.125f  // 64^-0.5

typedef __attribute__((ext_vector_type(16))) _Float16     v16h;
typedef __attribute__((ext_vector_type(8)))  float        v8f;
typedef __attribute__((ext_vector_type(4)))  unsigned int v4u;
typedef __attribute__((ext_vector_type(8)))  int          v8i;
typedef __attribute__((ext_vector_type(4)))  int          v4i;

// k-index striping for 16-bit A (and mirrored B^T) operands, per ISA 7.12.2:
// lane L holds row (L%16); kb = (L/16)*8; element i -> k = kb+i (i<8), kb+i+8 (i>=8)
__device__ __forceinline__ int kmap(int lane, int i) {
    int kb = (lane >> 4) << 3;
    return kb + i + ((i & 8) ? 8 : 0);
}

__device__ __forceinline__ v8f wmma_f16(v16h a, v16h b, v8f c) {
    // (neg_a, A, neg_b, B, c_mod, C, reuse_a, reuse_b)
    return __builtin_amdgcn_wmma_f32_16x16x32_f16(false, a, false, b, (short)0, c, false, false);
}

// ---------------------------------------------------------------------------
// Kernel 1: QKV projection.  C[r,c] = sum_k X[r,k] * W[c,k]   (y = x @ W.T)
// One wave computes TWO adjacent 16x16 output tiles, reusing the A operand
// (halves the x-stream traffic). 16 K-steps of v_wmma_f32_16x16x32_f16 x2.
// Output scattered into per-head layout [b][h][n][dh] as f16 (Q pre-scaled).
// grid: (1024 row-tiles, 4 col-groups, 3 {Q,K,V}), block: 128 (4 waves)
// ---------------------------------------------------------------------------
__global__ __launch_bounds__(128) void qkv_gemm_kernel(
    const float* __restrict__ x,
    const float* __restrict__ Wq, const float* __restrict__ Wk,
    const float* __restrict__ Wv,
    _Float16* __restrict__ qh, _Float16* __restrict__ kh,
    _Float16* __restrict__ vh)
{
    const int lane = threadIdx.x & 31;
    const int wvid = threadIdx.x >> 5;
    const int rt   = blockIdx.x;                      // 16384/16 row tiles
    const int ct0  = (blockIdx.y * 4 + wvid) * 2;     // first of 2 col tiles
    const int which = blockIdx.z;

    const float* W = (which == 0) ? Wq : (which == 1) ? Wk : Wv;
    _Float16* out  = (which == 0) ? qh : (which == 1) ? kh : vh;
    const float scale = (which == 0) ? SCALE_ : 1.0f;

    const int m     = lane & 15;
    const int row   = rt * 16 + m;                    // X row (A-matrix row)
    const int wrow0 = ct0 * 16 + (lane & 15);         // W row (output col), tile 0
    const int wrow1 = wrow0 + 16;                     // tile 1

    v8f acc0 = {}, acc1 = {};
    for (int ks = 0; ks < DIM_ / 32; ++ks) {
        v16h a, b0, b1;
#pragma unroll
        for (int i = 0; i < 16; ++i) {
            const int k = ks * 32 + kmap(lane, i);
            a[i]  = (_Float16)x[(size_t)row   * DIM_ + k];
            b0[i] = (_Float16)W[(size_t)wrow0 * DIM_ + k];
            b1[i] = (_Float16)W[(size_t)wrow1 * DIM_ + k];
        }
        acc0 = wmma_f16(a, b0, acc0);
        acc1 = wmma_f16(a, b1, acc1);
    }

    // Scatter C tiles into [b][h][n][dh] f16.  Tile rows share b (N%16==0).
    const int bb    = (rt * 16) / N_;
    const int nbase = (rt * 16) % N_;
#pragma unroll
    for (int t = 0; t < 2; ++t) {
        const v8f acc = t ? acc1 : acc0;
        const int c   = (ct0 + t) * 16 + (lane & 15);
        const int h   = c >> 6;
        const int dh  = c & 63;
#pragma unroll
        for (int v = 0; v < 8; ++v) {
            const int mm = v + ((lane >> 4) << 3);
            const int n  = nbase + mm;
            out[(((size_t)bb * H_ + h) * N_ + n) * DH_ + dh] =
                (_Float16)(acc[v] * scale);
        }
    }
}

// ---------------------------------------------------------------------------
// Kernel 2: attention for one (b, h, 16-row tile).
// Wave 0 kicks a TDM tensor_load_to_lds of the whole V head (1024x64 f16,
// 128 KB) at entry; the DMA overlaps Q@K^T and softmax, then all waves read
// V from LDS for attn@V.  LDS: V 128K + sim 64K + reductions (<320K/WGP).
// grid: (64 m-tiles, 128 b*h), block: 128 (4 waves)
// ---------------------------------------------------------------------------
__global__ __launch_bounds__(128) void attn_kernel(
    const _Float16* __restrict__ qh, const _Float16* __restrict__ kh,
    const _Float16* __restrict__ vh, const float* __restrict__ pos_bias,
    _Float16* __restrict__ oh)
{
    extern __shared__ char smemraw[];
    _Float16* vbuf = (_Float16*)smemraw;                       // [N_][DH_] 128KB
    float*    sim  = (float*)(smemraw + (size_t)N_ * DH_ * 2); // [16][N_]  64KB
    float*    red  = sim + 16 * N_;                            // [16][8]
    float*    invs = red + 16 * 8;                             // [16]

    const int lane  = threadIdx.x & 31;
    const int wvid  = threadIdx.x >> 5;
    const int mtile = blockIdx.x;       // 0..63
    const int bh    = blockIdx.y;       // 0..127
    const int bb    = bh >> 3;
    const int h     = bh & 7;
    const size_t head_off = ((size_t)bb * H_ + h) * (size_t)N_ * DH_;

    const int m = lane & 15;

    // --- TDM: async DMA the whole V head into LDS (wave 0 only).
    // D# per ISA 8.3/8.4: 2-D tensor, data_size=2B, tensor 64x1024 stride 64,
    // tile 64x1024, groups 2/3/4 zero (2-D), type=2.
    if (wvid == 0) {
        const uint64_t gaddr = (uint64_t)(uintptr_t)(vh + head_off);
        const unsigned ldsa  = (unsigned)(uintptr_t)vbuf; // low 32b = LDS addr
        v4u g0;
        g0[0] = 1u;                                   // count=1, user mode
        g0[1] = ldsa;                                 // lds_addr
        g0[2] = (unsigned)(gaddr & 0xffffffffu);      // global_addr[31:0]
        g0[3] = (unsigned)((gaddr >> 32) & 0x1ffffffu) | (2u << 30); // [56:32]|type=2
        v8i g1;
        g1[0] = (int)(1u << 16);       // workgroup_mask=0, data_size=1 (2B)
        g1[1] = (int)(64u << 16);      // tensor_dim0=64  (bits 79:48)
        g1[2] = (int)(1024u << 16);    // tensor_dim1=1024 (bits 111:80)
        g1[3] = (int)(64u << 16);      // tile_dim0=64    (bits 127:112)
        g1[4] = 1024;                  // tile_dim1=1024, tile_dim2=0
        g1[5] = 64;                    // tensor_dim0_stride=64 (bits 207:160)
        g1[6] = 0;
        g1[7] = 0;
        const v4i gz4 = {0, 0, 0, 0};
        const v8i gz8 = {0, 0, 0, 0, 0, 0, 0, 0};
        // 6-arg form (amdgpu-toolchain): (g0, g1, g2, g3, g4, cpol)
        __builtin_amdgcn_tensor_load_to_lds(g0, g1, gz4, gz4, gz8, 0);
    }

    // --- load Q tile rows (16 x 64 f16, pre-scaled) into A-operand registers
    v16h qa0, qa1;
    {
        const _Float16* qrow = qh + head_off + (size_t)(mtile * 16 + m) * DH_;
#pragma unroll
        for (int i = 0; i < 16; ++i) {
            const int k = kmap(lane, i);
            qa0[i] = qrow[k];
            qa1[i] = qrow[32 + k];
        }
    }

    // --- Phase A: sim = Q @ K^T + bias, this wave covers 16 column tiles
    for (int t = 0; t < 16; ++t) {
        const int ct = wvid * 16 + t;
        const int j  = ct * 16 + (lane & 15);        // sequence column
        // prefetch the bias line this tile will consume after the WMMAs
        if ((lane & 15) == 0)
            __builtin_prefetch(&pos_bias[((size_t)h * N_ + mtile * 16) * N_ + j], 0, 1);
        const _Float16* krow = kh + head_off + (size_t)j * DH_;
        v16h b0, b1;
#pragma unroll
        for (int i = 0; i < 16; ++i) {
            const int k = kmap(lane, i);
            b0[i] = krow[k];
            b1[i] = krow[32 + k];
        }
        v8f acc = {};
        acc = wmma_f16(qa0, b0, acc);
        acc = wmma_f16(qa1, b1, acc);
#pragma unroll
        for (int v = 0; v < 8; ++v) {
            const int mm   = v + ((lane >> 4) << 3);
            const int irow = mtile * 16 + mm;
            sim[mm * N_ + j] =
                acc[v] + pos_bias[((size_t)h * N_ + irow) * N_ + j];
        }
    }
    __syncthreads();

    // --- Phase B: softmax over each of 16 rows; 8 threads per row
    {
        const int r   = threadIdx.x >> 3;
        const int sub = threadIdx.x & 7;
        float* rowp = sim + r * N_ + sub * 128;

        float lmax = -3.4e38f;
        for (int j = 0; j < 128; ++j) lmax = fmaxf(lmax, rowp[j]);
        red[r * 8 + sub] = lmax;
        __syncthreads();
        float rmax = red[r * 8];
#pragma unroll
        for (int q = 1; q < 8; ++q) rmax = fmaxf(rmax, red[r * 8 + q]);
        __syncthreads();

        float lsum = 0.f;
        for (int j = 0; j < 128; ++j) {
            const float e = __expf(rowp[j] - rmax);
            rowp[j] = e;
            lsum += e;
        }
        red[r * 8 + sub] = lsum;
        __syncthreads();
        if (sub == 0) {
            float s = 0.f;
#pragma unroll
            for (int q = 0; q < 8; ++q) s += red[r * 8 + q];
            invs[r] = 1.f / s;
        }
    }
    // V DMA must be complete before phase C; wave 0 owns TENSORcnt, the
    // barrier releases everyone after the wait.
    if (wvid == 0) __builtin_amdgcn_s_wait_tensorcnt(0);
    __syncthreads();

    // --- Phase C: out(16 x 16 cols of DH per wave) = attn @ V (V from LDS)
    {
        const int dhb = wvid * 16;
        const int dhc = dhb + (lane & 15);
        v8f acc = {};
        for (int kt = 0; kt < N_ / 32; ++kt) {
            v16h a, b;
#pragma unroll
            for (int i = 0; i < 16; ++i) {
                const int k = kt * 32 + kmap(lane, i);
                a[i] = (_Float16)(sim[m * N_ + k] * invs[m]);   // normalized attn
                b[i] = vbuf[k * DH_ + dhc];
            }
            acc = wmma_f16(a, b, acc);
        }
#pragma unroll
        for (int v = 0; v < 8; ++v) {
            const int mm = v + ((lane >> 4) << 3);
            const int n  = mtile * 16 + mm;
            oh[((size_t)bb * N_ + n) * DIM_ + h * DH_ + dhb + (lane & 15)] =
                (_Float16)acc[v];
        }
    }
}

// ---------------------------------------------------------------------------
// Kernel 3: output projection.  out[r,c] = sum_k AO[r,k] * Wo[c,k], f32 out.
// Two 16x16 tiles per wave (A-operand reuse).  grid: (1024, 4), block 128.
// ---------------------------------------------------------------------------
__global__ __launch_bounds__(128) void out_gemm_kernel(
    const _Float16* __restrict__ oh, const float* __restrict__ Wo,
    float* __restrict__ out)
{
    const int lane = threadIdx.x & 31;
    const int wvid = threadIdx.x >> 5;
    const int rt   = blockIdx.x;
    const int ct0  = (blockIdx.y * 4 + wvid) * 2;

    const int m     = lane & 15;
    const int row   = rt * 16 + m;
    const int wrow0 = ct0 * 16 + (lane & 15);
    const int wrow1 = wrow0 + 16;

    v8f acc0 = {}, acc1 = {};
    for (int ks = 0; ks < DIM_ / 32; ++ks) {
        v16h a, b0, b1;
#pragma unroll
        for (int i = 0; i < 16; ++i) {
            const int k = ks * 32 + kmap(lane, i);
            a[i]  = oh[(size_t)row * DIM_ + k];
            b0[i] = (_Float16)Wo[(size_t)wrow0 * DIM_ + k];
            b1[i] = (_Float16)Wo[(size_t)wrow1 * DIM_ + k];
        }
        acc0 = wmma_f16(a, b0, acc0);
        acc1 = wmma_f16(a, b1, acc1);
    }
#pragma unroll
    for (int t = 0; t < 2; ++t) {
        const v8f acc = t ? acc1 : acc0;
#pragma unroll
        for (int v = 0; v < 8; ++v) {
            const int mm = v + ((lane >> 4) << 3);
            out[(size_t)(rt * 16 + mm) * DIM_ + (ct0 + t) * 16 + (lane & 15)] =
                acc[v];
        }
    }
}

// ---------------------------------------------------------------------------
extern "C" void kernel_launch(void* const* d_in, const int* in_sizes, int n_in,
                              void* d_out, int out_size, void* d_ws, size_t ws_size,
                              hipStream_t stream) {
    (void)in_sizes; (void)n_in; (void)out_size; (void)ws_size;
    const float* x        = (const float*)d_in[0];
    const float* pos_bias = (const float*)d_in[1];
    const float* Wq       = (const float*)d_in[2];
    const float* Wk       = (const float*)d_in[3];
    const float* Wv       = (const float*)d_in[4];
    const float* Wo       = (const float*)d_in[5];

    const size_t head_elems = (size_t)B_ * H_ * N_ * DH_;   // 8 Mi f16 = 16 MB
    _Float16* qh = (_Float16*)d_ws;
    _Float16* kh = qh + head_elems;
    _Float16* vh = kh + head_elems;
    _Float16* oh = vh + head_elems;                          // [B][N][DIM] f16

    // 1) Q/K/V projections (Q pre-scaled by DH^-0.5)
    qkv_gemm_kernel<<<dim3((B_ * N_) / 16, DIM_ / 16 / 8, 3), 128, 0, stream>>>(
        x, Wq, Wk, Wv, qh, kh, vh);

    // 2) attention per (b, h, 16-row tile); V staged to LDS by TDM
    const size_t shmem = (size_t)N_ * DH_ * 2                 // V f16
                       + (size_t)(16 * N_ + 16 * 8 + 16) * sizeof(float);
    attn_kernel<<<dim3(N_ / 16, B_ * H_), 128, shmem, stream>>>(
        qh, kh, vh, pos_bias, oh);

    // 3) output projection -> f32
    out_gemm_kernel<<<dim3((B_ * N_) / 16, DIM_ / 16 / 8), 128, 0, stream>>>(
        oh, Wo, (float*)d_out);
}
